// DecodeBlock_10488310136845
// MI455X (gfx1250) — matrix operand order repaired and checked by reading, not compiled, not run on hardware
//
#include <hip/hip_runtime.h>

// ---------------------------------------------------------------------------
// Problem constants (from reference): B=4, S=1024, E=512, H=8, D=64
// ---------------------------------------------------------------------------
#define BB 4
#define SS 1024
#define EE 512
#define HH 8
#define DD 64
#define NTOK (BB * SS)          // 4096 token rows
#define DSF 0.8f
#define EPS 1e-6f

typedef __bf16 v16bf __attribute__((ext_vector_type(16)));
typedef float  v8f   __attribute__((ext_vector_type(8)));

static __device__ inline v8f wmma_bf16(v16bf a, v16bf b, v8f c) {
  // D = A(16x32 bf16) * B(32x16 bf16) + C(16x16 f32)
  return __builtin_amdgcn_wmma_f32_16x16x32_bf16(
      false, a, false, b, (short)0, c, false, false);
}

static __device__ inline v8f zero8() {
  v8f z;
#pragma unroll
  for (int i = 0; i < 8; ++i) z[i] = 0.0f;
  return z;
}

static __device__ inline float silu_f(float x) {
  return x / (1.0f + __expf(-x));
}

// Async 16-byte global -> LDS copy (per-lane addresses), ASYNCcnt-tracked.
static __device__ inline void async_ld16(unsigned lds_off, const __bf16* g) {
  asm volatile("global_load_async_to_lds_b128 %0, %1, off"
               :: "v"(lds_off), "v"(g) : "memory");
}

template <int N>
static __device__ inline void wait_asynccnt() {
  asm volatile("s_wait_asynccnt %0" :: "i"(N) : "memory");
}

static __device__ inline unsigned lds_addr(const void* p) {
  return (unsigned)(size_t)p;  // LDS offset lives in the low 32 bits
}

// ---------------------------------------------------------------------------
// fp32 -> bf16 conversion (linear)
// ---------------------------------------------------------------------------
__global__ void f32_to_bf16_kernel(const float* __restrict__ in,
                                   __bf16* __restrict__ out, int n) {
  for (int i = blockIdx.x * 256 + threadIdx.x; i < n; i += gridDim.x * 256)
    out[i] = (__bf16)in[i];
}

// Weight prep: Wt[n][k] = (bf16) W[k][n]  (512x512)
__global__ void wt_bf16_kernel(const float* __restrict__ w,
                               __bf16* __restrict__ wt) {
  for (int idx = blockIdx.x * 256 + threadIdx.x; idx < EE * EE;
       idx += gridDim.x * 256) {
    const int n = idx >> 9, k = idx & (EE - 1);
    wt[idx] = (__bf16)w[k * EE + n];
  }
}

// ---------------------------------------------------------------------------
// GEMM: C = scale * A[4096x512] @ W[512x512]; A is bf16, W pre-transposed
// bf16 [n][k]. Double-buffered LDS tiles filled with async-to-LDS copies.
// Block = 256 threads (8 waves), tile 128x128, K-step 32.
// Writes fp32 C and a bf16 mirror for downstream matrix consumers.
// ---------------------------------------------------------------------------
__global__ __launch_bounds__(256) void gemm_bf16_kernel(
    const __bf16* __restrict__ Abf, const __bf16* __restrict__ Wt,
    float* __restrict__ C, __bf16* __restrict__ Cbf, float scale) {
  constexpr int K = EE, N = EE;
  __shared__ __align__(16) __bf16 lds_a[2][128 * 32];   // [m][k]
  __shared__ __align__(16) __bf16 lds_bt[2][128 * 32];  // [n][k]

  const int tid  = threadIdx.x;
  const int lane = tid & 31, wave = tid >> 5;
  const int l16  = lane & 15, hlf = lane >> 4;
  const int m0   = blockIdx.x * 128, n0 = blockIdx.y * 128;
  const int wm   = (wave & 1) * 64, wn = (wave >> 1) * 32;

  v8f acc[4][2];
#pragma unroll
  for (int mt = 0; mt < 4; ++mt)
#pragma unroll
    for (int nt = 0; nt < 2; ++nt) acc[mt][nt] = zero8();

  auto issue = [&](int kt, int bufi) {
    const int k0 = kt * 32;
    int c = tid;
#pragma unroll
    for (int it = 0; it < 2; ++it, c += 256) {  // A tile: 512 x 16B chunks
      const int row = c >> 2, part = c & 3;
      async_ld16(lds_addr(&lds_a[bufi][row * 32 + part * 8]),
                 Abf + (size_t)(m0 + row) * K + k0 + part * 8);
    }
    c = tid;
#pragma unroll
    for (int it = 0; it < 2; ++it, c += 256) {  // Wt tile (already [n][k])
      const int row = c >> 2, part = c & 3;
      async_ld16(lds_addr(&lds_bt[bufi][row * 32 + part * 8]),
                 Wt + (size_t)(n0 + row) * K + k0 + part * 8);
    }
  };

  issue(0, 0);
  for (int kt = 0; kt < K / 32; ++kt) {
    const int cur = kt & 1;
    if (kt + 1 < K / 32) {
      issue(kt + 1, cur ^ 1);
      wait_asynccnt<4>();   // tile kt complete; next tile stays in flight
    } else {
      wait_asynccnt<0>();
    }
    __syncthreads();

    v16bf afr[4];
#pragma unroll
    for (int mt = 0; mt < 4; ++mt) {
      const __bf16* p = &lds_a[cur][(wm + mt * 16 + l16) * 32];
#pragma unroll
      for (int e = 0; e < 16; ++e) {
        const int kk = e + hlf * 16;
        afr[mt][e] = p[kk];
      }
    }
#pragma unroll
    for (int nt = 0; nt < 2; ++nt) {
      v16bf bfr;
      const __bf16* p = &lds_bt[cur][(wn + nt * 16 + l16) * 32 + hlf * 16];
#pragma unroll
      for (int e = 0; e < 16; ++e) bfr[e] = p[e];
#pragma unroll
      for (int mt = 0; mt < 4; ++mt)
        acc[mt][nt] = wmma_bf16(afr[mt], bfr, acc[mt][nt]);
    }
    __syncthreads();  // all waves done with buf[cur] before it is refilled
  }

#pragma unroll
  for (int mt = 0; mt < 4; ++mt)
#pragma unroll
    for (int nt = 0; nt < 2; ++nt)
#pragma unroll
      for (int r = 0; r < 8; ++r) {
        const int row = m0 + wm + mt * 16 + r + hlf * 8;
        const int col = n0 + wn + nt * 16 + l16;
        const float val = scale * acc[mt][nt][r];
        C[(size_t)row * N + col]   = val;
        Cbf[(size_t)row * N + col] = (__bf16)val;
      }
}

// ---------------------------------------------------------------------------
// Segment scan: seg[b,s] = cumsum(dones[b,s]) (inclusive).
// ---------------------------------------------------------------------------
__global__ void seg_scan_kernel(const int* __restrict__ dones,
                                int* __restrict__ seg) {
  const int b = blockIdx.x;
  if (threadIdx.x != 0) return;
  int c = 0;
  for (int s = 0; s < SS; ++s) {
    c += (dones[b * SS + s] != 0);
    seg[b * SS + s] = c;
  }
}

// ---------------------------------------------------------------------------
// Retention core: ret[b,i,h,:] = sum_j decay(i,j) (q_i . k_j) v_j + xi_i q_i H
// Block = (i-tile of 128, head, batch); 8 waves x 16 rows. q/k tiles arrive
// via async-to-LDS; decayed scores bounce C-layout -> LDS -> A-layout.
// ---------------------------------------------------------------------------
__global__ __launch_bounds__(256) void retention_core_kernel(
    const __bf16* __restrict__ q, const __bf16* __restrict__ k,
    const __bf16* __restrict__ v, const float* __restrict__ h0,
    const int* __restrict__ seg, const int* __restrict__ ts,
    float* __restrict__ ret) {
  const int i0 = blockIdx.x * 128;
  const int hh = blockIdx.y;
  const int b  = blockIdx.z;
  const float kap = (1.0f - exp2f(-5.0f - (float)hh)) * DSF;
  const float lk  = __logf(kap);

  __shared__ __align__(16) __bf16 lq[128 * 64];   // [i][d]
  __shared__ __align__(16) __bf16 lht[64 * 64];   // [v][d] (h transposed)
  __shared__ __align__(16) __bf16 lkt[32 * 64];   // [jj][d]
  __shared__ __align__(16) __bf16 lvt[64 * 32];   // [d][jj]
  __shared__ __align__(16) __bf16 lsc[8 * 16 * 32];
  __shared__ float  tsi[128];
  __shared__ int    sgi[128];
  __shared__ float  tsj[32];
  __shared__ int    sgj[32];

  const int tid  = threadIdx.x;
  const int lane = tid & 31, wave = tid >> 5;
  const int l16  = lane & 15, hlf = lane >> 4;

  {  // q tile: 128x64 bf16 = 16 KB, async linear copy (4 x b128 per lane)
    int c = tid;
#pragma unroll
    for (int it = 0; it < 4; ++it, c += 256) {
      const int row = c >> 3, part = c & 7;
      async_ld16(lds_addr(&lq[row * 64 + part * 8]),
                 q + (size_t)(b * SS + i0 + row) * EE + hh * DD + part * 8);
    }
  }
  for (int idx = tid; idx < 64 * 64; idx += 256) {  // h transpose (manual)
    const int d = idx >> 6, vv = idx & 63;
    lht[vv * 64 + d] = (__bf16)h0[((size_t)(b * HH + hh) * DD + d) * DD + vv];
  }
  for (int idx = tid; idx < 128; idx += 256) {
    tsi[idx] = (float)ts[b * SS + i0 + idx];
    sgi[idx] = seg[b * SS + i0 + idx];
  }
  wait_asynccnt<0>();
  __syncthreads();

  // q A-fragments (rows wave*16 .. wave*16+15), held in regs for whole block
  v16bf aq[2];
#pragma unroll
  for (int ks = 0; ks < 2; ++ks) {
    const __bf16* p = &lq[(wave * 16 + l16) * 64 + ks * 32];
#pragma unroll
    for (int e = 0; e < 16; ++e) {
      const int kk = e + hlf * 16;
      aq[ks][e] = p[kk];
    }
  }

  v8f accd[4];
#pragma unroll
  for (int nt = 0; nt < 4; ++nt) accd[nt] = zero8();

  for (int j0 = 0; j0 < i0 + 128; j0 += 32) {
    __syncthreads();
    {  // k tile: 32x64 bf16 = 4 KB, async linear copy
      const int jj = tid >> 3, part = tid & 7;
      async_ld16(lds_addr(&lkt[jj * 64 + part * 8]),
                 k + (size_t)(b * SS + j0 + jj) * EE + hh * DD + part * 8);
    }
    for (int idx = tid; idx < 32 * 64; idx += 256) {  // v transpose (manual)
      const int jj = idx >> 6, d = idx & 63;
      lvt[d * 32 + jj] = v[(size_t)(b * SS + j0 + jj) * EE + hh * DD + d];
    }
    if (tid < 32) {
      tsj[tid] = (float)ts[b * SS + j0 + tid];
      sgj[tid] = seg[b * SS + j0 + tid];
    }
    wait_asynccnt<0>();
    __syncthreads();

    // scores: 16 x 32 per wave
    v8f sc[2];
#pragma unroll
    for (int t = 0; t < 2; ++t) {
      sc[t] = zero8();
#pragma unroll
      for (int ks = 0; ks < 2; ++ks) {
        v16bf bk;
        const __bf16* p = &lkt[(t * 16 + l16) * 64 + ks * 32 + hlf * 16];
#pragma unroll
        for (int e = 0; e < 16; ++e) bk[e] = p[e];
        sc[t] = wmma_bf16(aq[ks], bk, sc[t]);
      }
    }

    // decay + causal/segment mask; stage to LDS as bf16
#pragma unroll
    for (int t = 0; t < 2; ++t)
#pragma unroll
      for (int r = 0; r < 8; ++r) {
        const int il = wave * 16 + r + hlf * 8;
        const int jl = t * 16 + l16;
        const int gi = i0 + il, gj = j0 + jl;
        float s = sc[t][r];
        const float td = tsi[il] - tsj[jl];
        const bool m = (gi >= gj) && (sgi[il] == sgj[jl]);
        s = m ? s * __expf(td * lk) : 0.0f;
        lsc[wave * 512 + (r + hlf * 8) * 32 + jl] = (__bf16)s;
      }

    // reload decayed scores as A-fragment (same wave; DScnt ordering)
    v16bf as_;
    {
      const __bf16* p = &lsc[wave * 512 + l16 * 32];
#pragma unroll
      for (int e = 0; e < 16; ++e) {
        const int kk = e + hlf * 16;
        as_[e] = p[kk];
      }
    }
#pragma unroll
    for (int nt = 0; nt < 4; ++nt) {
      v16bf bv;
      const __bf16* p = &lvt[(nt * 16 + l16) * 32 + hlf * 16];
#pragma unroll
      for (int e = 0; e < 16; ++e) bv[e] = p[e];
      accd[nt] = wmma_bf16(as_, bv, accd[nt]);
    }
  }

  // cross term: accd += (xi_i * q_i) @ H
  {
    const int il = wave * 16 + l16;
    const float xi = (sgi[il] == 0) ? __expf((tsi[il] + 1.0f) * lk) : 0.0f;
#pragma unroll
    for (int ks = 0; ks < 2; ++ks) {
      v16bf ax;
      const __bf16* p = &lq[il * 64 + ks * 32];
#pragma unroll
      for (int e = 0; e < 16; ++e) {
        const int kk = e + hlf * 16;
        ax[e] = (__bf16)((float)p[kk] * xi);
      }
#pragma unroll
      for (int nt = 0; nt < 4; ++nt) {
        v16bf bh;
        const __bf16* ph = &lht[(nt * 16 + l16) * 64 + ks * 32 + hlf * 16];
#pragma unroll
        for (int e = 0; e < 16; ++e) bh[e] = ph[e];
        accd[nt] = wmma_bf16(ax, bh, accd[nt]);
      }
    }
  }

#pragma unroll
  for (int nt = 0; nt < 4; ++nt)
#pragma unroll
    for (int r = 0; r < 8; ++r) {
      const int i = i0 + wave * 16 + r + hlf * 8;
      const int d = nt * 16 + l16;
      ret[(size_t)(b * SS + i) * EE + hh * DD + d] = accd[nt][r];
    }
}

// ---------------------------------------------------------------------------
// State update: h_new[d,v] = sum_j eta_j k[j,d] v[j,v] + h0[d,v]*fade
// Block per (head, batch); 8 waves over 64x64 output; K = S = 1024.
// ---------------------------------------------------------------------------
__global__ __launch_bounds__(256) void hstate_kernel(
    const __bf16* __restrict__ k, const __bf16* __restrict__ v,
    const float* __restrict__ h0, const int* __restrict__ seg,
    const int* __restrict__ ts, float* __restrict__ hout) {
  const int hh = blockIdx.x, b = blockIdx.y;
  const float kap  = (1.0f - exp2f(-5.0f - (float)hh)) * DSF;
  const float lk   = __logf(kap);
  const int   segL = seg[b * SS + SS - 1];
  const float tsL  = (float)ts[b * SS + SS - 1];

  __shared__ __align__(16) __bf16 lkt[64 * 32];  // [d][jj], eta-scaled
  __shared__ __align__(16) __bf16 lvt[64 * 32];  // [v][jj]

  const int tid  = threadIdx.x;
  const int lane = tid & 31, wave = tid >> 5;
  const int l16  = lane & 15, hlf = lane >> 4;
  const int d0   = (wave & 3) * 16, v0 = (wave >> 2) * 32;

  v8f acc[2];
#pragma unroll
  for (int nt = 0; nt < 2; ++nt) acc[nt] = zero8();

  for (int j0 = 0; j0 < SS; j0 += 32) {
    __syncthreads();
    for (int idx = tid; idx < 64 * 32; idx += 256) {
      const int d = idx >> 5, jj = idx & 31;
      const int j = j0 + jj;
      const size_t g = (size_t)(b * SS + j) * EE + hh * DD + d;
      const float eta = (seg[b * SS + j] == segL)
                            ? __expf((tsL - (float)ts[b * SS + j]) * lk)
                            : 0.0f;
      lkt[idx] = (__bf16)((float)k[g] * eta);
      lvt[idx] = v[g];
    }
    __syncthreads();

    v16bf ak;
    {
      const __bf16* p = &lkt[(d0 + l16) * 32];
#pragma unroll
      for (int e = 0; e < 16; ++e) {
        const int kk = e + hlf * 16;
        ak[e] = p[kk];
      }
    }
#pragma unroll
    for (int nt = 0; nt < 2; ++nt) {
      v16bf bv;
      const __bf16* p = &lvt[(v0 + nt * 16 + l16) * 32 + hlf * 16];
#pragma unroll
      for (int e = 0; e < 16; ++e) bv[e] = p[e];
      acc[nt] = wmma_bf16(ak, bv, acc[nt]);
    }
  }

  const float fade = (segL == 0) ? __expf((tsL + 1.0f) * lk) : 0.0f;
#pragma unroll
  for (int nt = 0; nt < 2; ++nt)
#pragma unroll
    for (int r = 0; r < 8; ++r) {
      const int d  = d0 + r + hlf * 8;
      const int vv = v0 + nt * 16 + l16;
      const size_t o = ((size_t)(b * HH + hh) * DD + d) * DD + vv;
      hout[o] = acc[nt][r] + h0[o] * fade;
    }
}

// ---------------------------------------------------------------------------
// GroupNorm (per head, D=64) + SiLU gate; writes fp32 + bf16 mirror.
// ---------------------------------------------------------------------------
__global__ __launch_bounds__(256) void gn_gate_kernel(
    const float* __restrict__ ret, const float* __restrict__ g,
    const float* __restrict__ gns, const float* __restrict__ gnb,
    float* __restrict__ out, __bf16* __restrict__ outbf) {
  const int row = blockIdx.x;
  const int t = threadIdx.x, e0 = t * 2;
  const float x0 = ret[(size_t)row * EE + e0];
  const float x1 = ret[(size_t)row * EE + e0 + 1];
  float s = x0 + x1, q2 = x0 * x0 + x1 * x1;
#pragma unroll
  for (int off = 16; off >= 1; off >>= 1) {
    s  += __shfl_xor(s, off, 32);
    q2 += __shfl_xor(q2, off, 32);
  }
  const float mean = s * (1.0f / 64.0f);
  const float var  = q2 * (1.0f / 64.0f) - mean * mean;
  const float inv  = rsqrtf(var + EPS);
  const float y0 = (x0 - mean) * inv * gns[e0] + gnb[e0];
  const float y1 = (x1 - mean) * inv * gns[e0 + 1] + gnb[e0 + 1];
  const float g0 = g[(size_t)row * EE + e0];
  const float g1 = g[(size_t)row * EE + e0 + 1];
  const float o0 = y0 * silu_f(g0), o1 = y1 * silu_f(g1);
  out[(size_t)row * EE + e0]       = o0;
  out[(size_t)row * EE + e0 + 1]   = o1;
  outbf[(size_t)row * EE + e0]     = (__bf16)o0;
  outbf[(size_t)row * EE + e0 + 1] = (__bf16)o1;
}

// ---------------------------------------------------------------------------
// out = (a+b) * rsqrt(mean((a+b)^2) + eps) * sc   (RMS norm over E=512)
// ---------------------------------------------------------------------------
__global__ __launch_bounds__(256) void add_rms_kernel(
    const float* __restrict__ a, const float* __restrict__ b,
    const float* __restrict__ sc, float* __restrict__ out,
    __bf16* __restrict__ outbf) {
  __shared__ float red[8];
  const int row = blockIdx.x, t = threadIdx.x, e0 = t * 2;
  const float v0 = a[(size_t)row * EE + e0] + b[(size_t)row * EE + e0];
  const float v1 = a[(size_t)row * EE + e0 + 1] + b[(size_t)row * EE + e0 + 1];
  float p = v0 * v0 + v1 * v1;
#pragma unroll
  for (int off = 16; off >= 1; off >>= 1) p += __shfl_xor(p, off, 32);
  if ((t & 31) == 0) red[t >> 5] = p;
  __syncthreads();
  float tot = 0.0f;
#pragma unroll
  for (int w = 0; w < 8; ++w) tot += red[w];
  const float inv = rsqrtf(tot * (1.0f / (float)EE) + EPS);
  const float o0 = v0 * inv * sc[e0], o1 = v1 * inv * sc[e0 + 1];
  out[(size_t)row * EE + e0]     = o0;
  out[(size_t)row * EE + e0 + 1] = o1;
  if (outbf) {
    outbf[(size_t)row * EE + e0]     = (__bf16)o0;
    outbf[(size_t)row * EE + e0 + 1] = (__bf16)o1;
  }
}

__global__ void silu_mul_kernel(const float* __restrict__ a,
                                const float* __restrict__ b,
                                float* __restrict__ out,
                                __bf16* __restrict__ outbf, int n) {
  for (int i = blockIdx.x * blockDim.x + threadIdx.x; i < n;
       i += gridDim.x * blockDim.x) {
    const float o = silu_f(a[i]) * b[i];
    out[i]   = o;
    outbf[i] = (__bf16)o;
  }
}

// ---------------------------------------------------------------------------
// Host-side orchestration (stream-ordered; graph-capture safe).
// ---------------------------------------------------------------------------
extern "C" void kernel_launch(void* const* d_in, const int* in_sizes, int n_in,
                              void* d_out, int out_size, void* d_ws,
                              size_t ws_size, hipStream_t stream) {
  (void)in_sizes; (void)n_in; (void)out_size; (void)ws_size;

  const float* x     = (const float*)d_in[0];
  const float* obs   = (const float*)d_in[1];
  const float* hs1   = (const float*)d_in[2];
  const float* hs2   = (const float*)d_in[3];
  const int*   dones = (const int*)d_in[4];
  const int*   tsid  = (const int*)d_in[5];
  const float* ln1   = (const float*)d_in[6];
  const float* ln2   = (const float*)d_in[7];
  const float* ln3   = (const float*)d_in[8];
  const float* r1gns = (const float*)d_in[14];
  const float* r1gnb = (const float*)d_in[15];
  const float* r2gns = (const float*)d_in[21];
  const float* r2gnb = (const float*)d_in[22];

  // GEMM weight sources in use order: r1 q,k,v,g,o | r2 q,k,v,g,o | ffn l,g,o
  const float* wsrc[13] = {
      (const float*)d_in[9],  (const float*)d_in[10], (const float*)d_in[11],
      (const float*)d_in[12], (const float*)d_in[13], (const float*)d_in[16],
      (const float*)d_in[17], (const float*)d_in[18], (const float*)d_in[19],
      (const float*)d_in[20], (const float*)d_in[23], (const float*)d_in[24],
      (const float*)d_in[25]};

  float* out  = (float*)d_out;
  float* hs1o = out + (size_t)NTOK * EE;                 // 2,097,152
  float* hs2o = hs1o + (size_t)BB * HH * DD * DD;        // +131,072

  // Workspace: 5 fp32 buffers + 7 bf16 buffers + 13 bf16 weights + seg ints
  float* W = (float*)d_ws;
  const size_t BUF = (size_t)NTOK * EE;
  float* b0 = W;
  float* b1 = W + BUF;
  float* b2 = W + 2 * BUF;
  float* b3 = W + 3 * BUF;
  float* b4 = W + 4 * BUF;
  __bf16* hb  = (__bf16*)(W + 5 * BUF);
  __bf16* b0h = hb;
  __bf16* b1h = hb + BUF;
  __bf16* b2h = hb + 2 * BUF;
  __bf16* b3h = hb + 3 * BUF;
  __bf16* b4h = hb + 4 * BUF;
  __bf16* xh   = hb + 5 * BUF;
  __bf16* obsh = hb + 6 * BUF;
  __bf16* wth  = hb + 7 * BUF;                    // 13 x 512x512 bf16
  int* segp = (int*)(wth + (size_t)13 * EE * EE);

  __bf16* wt[13];
  for (int i = 0; i < 13; ++i) wt[i] = wth + (size_t)i * EE * EE;

  const dim3 blk(256);
  const dim3 gg(NTOK / 128, EE / 128);
  const dim3 gcore(SS / 128, HH, BB);
  const dim3 ghs(HH, BB);
  const float KS = 0.125f;  // D^{-1/2}

  // ---- prep: segments, bf16 activations, bf16 transposed weights ----
  seg_scan_kernel<<<BB, 32, 0, stream>>>(dones, segp);
  f32_to_bf16_kernel<<<1024, blk, 0, stream>>>(x, xh, (int)BUF);
  f32_to_bf16_kernel<<<1024, blk, 0, stream>>>(obs, obsh, (int)BUF);
  for (int i = 0; i < 13; ++i)
    wt_bf16_kernel<<<256, blk, 0, stream>>>(wsrc[i], wt[i]);

  // ---- retention 1 (q=k=v=x) ----
  gemm_bf16_kernel<<<gg, blk, 0, stream>>>(xh, wt[0], b0, b0h, 1.0f);
  gemm_bf16_kernel<<<gg, blk, 0, stream>>>(xh, wt[1], b1, b1h, KS);
  gemm_bf16_kernel<<<gg, blk, 0, stream>>>(xh, wt[2], b2, b2h, 1.0f);
  gemm_bf16_kernel<<<gg, blk, 0, stream>>>(xh, wt[3], b3, b3h, 1.0f);
  retention_core_kernel<<<gcore, blk, 0, stream>>>(b0h, b1h, b2h, hs1, segp,
                                                   tsid, b4);
  hstate_kernel<<<ghs, blk, 0, stream>>>(b1h, b2h, hs1, segp, tsid, hs1o);
  gn_gate_kernel<<<NTOK, blk, 0, stream>>>(b4, b3, r1gns, r1gnb, b0, b0h);
  gemm_bf16_kernel<<<gg, blk, 0, stream>>>(b0h, wt[4], b1, b1h, 1.0f);
  add_rms_kernel<<<NTOK, blk, 0, stream>>>(x, b1, ln1, b2, b2h);  // r

  // ---- retention 2 (q=obs, k=v=r) ----
  gemm_bf16_kernel<<<gg, blk, 0, stream>>>(obsh, wt[5], b0, b0h, 1.0f);
  gemm_bf16_kernel<<<gg, blk, 0, stream>>>(b2h, wt[6], b1, b1h, KS);
  gemm_bf16_kernel<<<gg, blk, 0, stream>>>(b2h, wt[7], b3, b3h, 1.0f);
  gemm_bf16_kernel<<<gg, blk, 0, stream>>>(obsh, wt[8], b4, b4h, 1.0f);
  retention_core_kernel<<<gcore, blk, 0, stream>>>(b0h, b1h, b3h, hs2, segp,
                                                   tsid, b2);
  hstate_kernel<<<ghs, blk, 0, stream>>>(b1h, b3h, hs2, segp, tsid, hs2o);
  gn_gate_kernel<<<NTOK, blk, 0, stream>>>(b2, b4, r2gns, r2gnb, b0, b0h);
  gemm_bf16_kernel<<<gg, blk, 0, stream>>>(b0h, wt[9], b1, b1h, 1.0f);
  add_rms_kernel<<<NTOK, blk, 0, stream>>>(obs, b1, ln2, b2, b2h);  // y

  // ---- FFN ----
  gemm_bf16_kernel<<<gg, blk, 0, stream>>>(b2h, wt[11], b0, b0h, 1.0f);
  gemm_bf16_kernel<<<gg, blk, 0, stream>>>(b2h, wt[10], b1, b1h, 1.0f);
  silu_mul_kernel<<<2048, blk, 0, stream>>>(b0, b1, b3, b3h, (int)BUF);
  gemm_bf16_kernel<<<gg, blk, 0, stream>>>(b3h, wt[12], b0, b0h, 1.0f);
  add_rms_kernel<<<NTOK, blk, 0, stream>>>(b2, b0, ln3, out, nullptr);
}
